// MessageFunction_4913442586652
// MI455X (gfx1250) — matrix operand order repaired and checked by reading, not compiled, not run on hardware
//
#include <hip/hip_runtime.h>
#include <hip/hip_bf16.h>

typedef __attribute__((ext_vector_type(16))) _Float16 v16h;
typedef __attribute__((ext_vector_type(8)))  _Float16 v8h;
typedef __attribute__((ext_vector_type(8)))  float    v8f;

#define E_TOT   32768      // B*N*N = 8*64*64
#define TILES   (E_TOT/16) // 2048 M-tiles of 16 edge rows

// Padded LDS leading dims (f16 units): +8 f16 per row => per-lane bank stride
// of 4 (mod 64) for fragment loads -> conflict-free across a 16-lane half.
#define LD1P  40    // W1T rows of 32
#define LD2P  136   // W2T rows of 128
#define LD3P  264   // W3T rows of 256
#define LD4P  136   // W4rT rows of 128
#define LDA_P 136   // staging bufA rows of 128
#define LDB_P 264   // staging bufB rows of 256

// ---------------- prep: transpose weights to [Nout][K] f16 ----------------
__global__ void k_transpose_f16(const float* __restrict__ in, _Float16* __restrict__ out,
                                int K, int N) {
    int idx = blockIdx.x * blockDim.x + threadIdx.x;
    if (idx >= K * N) return;
    int n = idx / K, k = idx - n * K;
    out[idx] = (_Float16)in[k * N + n];   // out[n][k] = in[k][n]
}

// ---------------- prep: W4r[k][o] = sum_i W4[k][o*64+i]; stored transposed [64][128] ----
__global__ void k_reduce_w4(const float* __restrict__ W4, const float* __restrict__ b4,
                            _Float16* __restrict__ W4rT, float* __restrict__ b4r) {
    int idx = blockIdx.x * blockDim.x + threadIdx.x;
    if (idx < 64 * 128) {
        int k = idx & 127;
        int o = idx >> 7;
        const float* src = W4 + (size_t)k * 4096 + o * 64;
        float s = 0.f;
        #pragma unroll 8
        for (int i = 0; i < 64; ++i) s += src[i];
        W4rT[o * 128 + k] = (_Float16)s;
    }
    if (idx < 64) {
        const float* src = b4 + idx * 64;
        float s = 0.f;
        for (int i = 0; i < 64; ++i) s += src[i];
        b4r[idx] = s;
    }
}

// ---------------- fragment helpers ----------------
__device__ __forceinline__ v16h load_a_lds(const _Float16* base, int ld, int lm, int half, int kbase) {
    // A frag 16x32 f16: lane holds row lm; 16B chunks at K = kbase + half*16 and +8
    const _Float16* r = base + lm * ld + kbase + half * 16;
    union { v16h v; v8h h[2]; } u;
    u.h[0] = *(const v8h*)(r);
    u.h[1] = *(const v8h*)(r + 8);
    return u.v;
}

__device__ __forceinline__ v16h load_b_lds(const _Float16* base, int off) {
    // B frag 32x16 f16 from transposed weights: lane holds col, 16 contiguous K
    union { v16h v; v8h h[2]; } u;
    u.h[0] = *(const v8h*)(base + off);
    u.h[1] = *(const v8h*)(base + off + 8);
    return u.v;
}

__device__ __forceinline__ float relu(float x) { return x > 0.f ? x : 0.f; }

// ---------------- fused edge-MLP kernel ----------------
__launch_bounds__(128)
__global__ void k_ecmlp_fused(const float* __restrict__ e_vw, const float* __restrict__ hw,
                              const _Float16* __restrict__ w1t, const _Float16* __restrict__ w2t,
                              const _Float16* __restrict__ w3t, const _Float16* __restrict__ w4t,
                              const float* __restrict__ b1, const float* __restrict__ b2,
                              const float* __restrict__ b3, const float* __restrict__ b4r,
                              float* __restrict__ out) {
    extern __shared__ __align__(32) char smem[];
    _Float16* lw1 = (_Float16*)smem;          // 128 x LD1P =  5120 f16
    _Float16* lw2 = lw1 + 128 * LD1P;         // 256 x LD2P = 34816 f16
    _Float16* lw3 = lw2 + 256 * LD2P;         // 128 x LD3P = 33792 f16
    _Float16* lw4 = lw3 + 128 * LD3P;         //  64 x LD4P =  8704 f16
    float*    lb1 = (float*)(lw4 + 64 * LD4P);
    float*    lb2 = lb1 + 128;
    float*    lb3 = lb2 + 256;
    float*    lb4 = lb3 + 128;
    _Float16* stage = (_Float16*)(lb4 + 64);  // per-wave: 16xLDA_P + 16xLDB_P f16

    const int tid  = threadIdx.x;
    const int nthr = blockDim.x;

    // Cooperative weight load into padded LDS rows (16B chunks)
    for (int i = tid; i <  512; i += nthr) { int r = i >> 2, c = i & 3;
        ((uint4*)(lw1 + r * LD1P))[c] = ((const uint4*)(w1t + r * 32))[c]; }
    for (int i = tid; i < 4096; i += nthr) { int r = i >> 4, c = i & 15;
        ((uint4*)(lw2 + r * LD2P))[c] = ((const uint4*)(w2t + r * 128))[c]; }
    for (int i = tid; i < 4096; i += nthr) { int r = i >> 5, c = i & 31;
        ((uint4*)(lw3 + r * LD3P))[c] = ((const uint4*)(w3t + r * 256))[c]; }
    for (int i = tid; i < 1024; i += nthr) { int r = i >> 4, c = i & 15;
        ((uint4*)(lw4 + r * LD4P))[c] = ((const uint4*)(w4t + r * 128))[c]; }
    for (int i = tid; i <  128; i += nthr) lb1[i] = b1[i];
    for (int i = tid; i <  256; i += nthr) lb2[i] = b2[i];
    for (int i = tid; i <  128; i += nthr) lb3[i] = b3[i];
    for (int i = tid; i <   64; i += nthr) lb4[i] = b4r[i];
    __syncthreads();

    const int wave = tid >> 5;
    const int lane = tid & 31;
    const int lm   = lane & 15;   // row (A) / col (B,C,D) within tile
    const int half = lane >> 4;   // lane-half select

    _Float16* bufA = stage + wave * (16 * LDA_P + 16 * LDB_P); // 16 x 128 (padded)
    _Float16* bufB = bufA + 16 * LDA_P;                        // 16 x 256 (padded)

    const int wavesPerGrid = gridDim.x * (blockDim.x >> 5);
    for (int t = blockIdx.x * (blockDim.x >> 5) + wave; t < TILES; t += wavesPerGrid) {
        const int row_base = t * 16;

        // ---- layer 1: [16x32] @ W1[32x128] + b1, ReLU -> bufA ----
        v16h a0;
        {
            const float* r = e_vw + (size_t)(row_base + lm) * 32 + half * 16;
            #pragma unroll
            for (int i = 0; i < 8; ++i) a0[i] = (_Float16)r[i];
            #pragma unroll
            for (int i = 0; i < 8; ++i) a0[8 + i] = (_Float16)r[8 + i];
        }
        for (int j = 0; j < 8; ++j) {
            const int n = j * 16 + lm;
            const float bb = lb1[n];
            v8f c = {bb, bb, bb, bb, bb, bb, bb, bb};
            v16h bf = load_b_lds(lw1, n * LD1P + half * 16);
            c = __builtin_amdgcn_wmma_f32_16x16x32_f16(false, a0, false, bf, (short)0, c, false, false);
            #pragma unroll
            for (int v = 0; v < 8; ++v)
                bufA[(half * 8 + v) * LDA_P + n] = (_Float16)relu(c[v]);
        }

        v16h a[8];

        // ---- layer 2: [16x128] @ W2[128x256] + b2, ReLU -> bufB ----
        #pragma unroll
        for (int kc = 0; kc < 4; ++kc) a[kc] = load_a_lds(bufA, LDA_P, lm, half, kc * 32);
        for (int j = 0; j < 16; ++j) {
            const int n = j * 16 + lm;
            const float bb = lb2[n];
            v8f c = {bb, bb, bb, bb, bb, bb, bb, bb};
            #pragma unroll
            for (int kc = 0; kc < 4; ++kc) {
                v16h bf = load_b_lds(lw2, n * LD2P + kc * 32 + half * 16);
                c = __builtin_amdgcn_wmma_f32_16x16x32_f16(false, a[kc], false, bf, (short)0, c, false, false);
            }
            #pragma unroll
            for (int v = 0; v < 8; ++v)
                bufB[(half * 8 + v) * LDB_P + n] = (_Float16)relu(c[v]);
        }

        // ---- layer 3: [16x256] @ W3[256x128] + b3, ReLU -> bufA ----
        #pragma unroll
        for (int kc = 0; kc < 8; ++kc) a[kc] = load_a_lds(bufB, LDB_P, lm, half, kc * 32);
        for (int j = 0; j < 8; ++j) {
            const int n = j * 16 + lm;
            const float bb = lb3[n];
            v8f c = {bb, bb, bb, bb, bb, bb, bb, bb};
            #pragma unroll
            for (int kc = 0; kc < 8; ++kc) {
                v16h bf = load_b_lds(lw3, n * LD3P + kc * 32 + half * 16);
                c = __builtin_amdgcn_wmma_f32_16x16x32_f16(false, a[kc], false, bf, (short)0, c, false, false);
            }
            #pragma unroll
            for (int v = 0; v < 8; ++v)
                bufA[(half * 8 + v) * LDA_P + n] = (_Float16)relu(c[v]);
        }

        // ---- layer 4 (pre-reduced): [16x128] @ W4r[128x64] + b4r, scale by hw.flat ----
        #pragma unroll
        for (int kc = 0; kc < 4; ++kc) a[kc] = load_a_lds(bufA, LDA_P, lm, half, kc * 32);
        for (int j = 0; j < 4; ++j) {
            const int n = j * 16 + lm;
            const float bb = lb4[n];
            v8f c = {bb, bb, bb, bb, bb, bb, bb, bb};
            #pragma unroll
            for (int kc = 0; kc < 4; ++kc) {
                v16h bf = load_b_lds(lw4, n * LD4P + kc * 32 + half * 16);
                c = __builtin_amdgcn_wmma_f32_16x16x32_f16(false, a[kc], false, bf, (short)0, c, false, false);
            }
            #pragma unroll
            for (int v = 0; v < 8; ++v) {
                const int m = row_base + half * 8 + v;
                out[(size_t)m * 64 + n] = hw[m] * c[v];
            }
        }
    }
}

extern "C" void kernel_launch(void* const* d_in, const int* in_sizes, int n_in,
                              void* d_out, int out_size, void* d_ws, size_t ws_size,
                              hipStream_t stream) {
    // setup_inputs order: h_v(0, unused), h_w(1), e_vw(2), W1(3), b1(4), W2(5), b2(6),
    //                     W3(7), b3(8), W4(9), b4(10)
    const float* h_w  = (const float*)d_in[1];
    const float* e_vw = (const float*)d_in[2];
    const float* W1   = (const float*)d_in[3];
    const float* b1   = (const float*)d_in[4];
    const float* W2   = (const float*)d_in[5];
    const float* b2   = (const float*)d_in[6];
    const float* W3   = (const float*)d_in[7];
    const float* b3   = (const float*)d_in[8];
    const float* W4   = (const float*)d_in[9];
    const float* b4   = (const float*)d_in[10];

    char* ws = (char*)d_ws;
    _Float16* w1t = (_Float16*)(ws);                            //  4096 f16 ->  8192 B
    _Float16* w2t = (_Float16*)(ws + 8192);                     // 32768 f16 -> 65536 B
    _Float16* w3t = (_Float16*)(ws + 8192 + 65536);             // 32768 f16 -> 65536 B
    _Float16* w4t = (_Float16*)(ws + 8192 + 65536 + 65536);     //  8192 f16 -> 16384 B
    float*    b4r = (float*)(ws + 8192 + 65536 + 65536 + 16384);

    k_transpose_f16<<<(32 * 128 + 255) / 256, 256, 0, stream>>>(W1, w1t, 32, 128);
    k_transpose_f16<<<(128 * 256 + 255) / 256, 256, 0, stream>>>(W2, w2t, 128, 256);
    k_transpose_f16<<<(256 * 128 + 255) / 256, 256, 0, stream>>>(W3, w3t, 256, 128);
    k_reduce_w4<<<32, 256, 0, stream>>>(W4, b4, w4t, b4r);

    // LDS bytes: weights 82432 f16 = 164864, biases 576 f32 = 2304,
    //            staging 4 waves * (16*136 + 16*264) f16 = 51200  -> 218368 total
    const size_t shmem = 218368;
    k_ecmlp_fused<<<TILES / 4, 128, shmem, stream>>>(e_vw, h_w, w1t, w2t, w3t, w4t,
                                                     b1, b2, b3, b4r, (float*)d_out);
}
